// VectorQuantizer_2645699854569
// MI455X (gfx1250) — matrix operand order repaired and checked by reading, not compiled, not run on hardware
//
#include <hip/hip_runtime.h>

// ---------------------------------------------------------------------------
// VectorQuantizer forward for MI455X (gfx1250, wave32, WMMA).
//
// Roofline: outputs ~302 MB -> ~13 us floor @ 23.3 TB/s. GEMM is 34.4 GFLOP;
// done as 3x bf16 WMMA (split hi/lo, f32 accum) = ~103 GFLOP, well under the
// matrix ceiling, so the kernel should be bound by the one-hot store stream.
// Codebook (2 MB bf16 hi+lo) stays in the 192 MB L2; each of 512 blocks
// re-reads it once -> ~1 GB of L2 traffic.
//
// Round-1 fix: unroll(1) on the k-loop to kill VGPR spills.
// Round-2 fix: nontemporal builtins need ext-vector types, not HIP float4.
// ---------------------------------------------------------------------------

#define KD    256      // code dimension D
#define NC    2048     // number of codes K
#define NROWS 32768    // 32 * 1024 flattened rows
#define MTILE 64       // rows per block

typedef __attribute__((ext_vector_type(16))) __bf16 v16bf;
typedef __attribute__((ext_vector_type(8)))  __bf16 v8bf;
typedef __attribute__((ext_vector_type(4)))  __bf16 v4bf;
typedef __attribute__((ext_vector_type(8)))  float  v8f;
typedef __attribute__((ext_vector_type(4)))  float  v4f;

__device__ __forceinline__ v8f wmma_bf16(v16bf a, v16bf b, v8f c) {
  // (neg_a, A, neg_b, B, c_mod, C, reuse_a, reuse_b)
  return __builtin_amdgcn_wmma_f32_16x16x32_bf16(false, a, false, b, (short)0, c,
                                                 false, false);
}

#define CAT16(lo8, hi8) \
  __builtin_shufflevector((lo8), (hi8), 0, 1, 2, 3, 4, 5, 6, 7, 8, 9, 10, 11, 12, 13, 14, 15)

// XOR swizzle on the K element index (8-element granularity) -> A-fragment
// ds_load_b128 across 16 rows hits 16 distinct 4-dword bank groups.
__device__ __forceinline__ int aswz(int row, int k) { return k ^ ((row & 15) << 3); }

union SmemU {
  struct {  // GEMM phase: 64 x 256 bf16 hi + lo  = 65536 B
    __bf16 a_hi[MTILE][KD];
    __bf16 a_lo[MTILE][KD];
  } g;
  struct {  // reduction phase (reused after barrier): 4352 B
    float bestv[MTILE][8];
    int   besti[MTILE][8];
    int   idxf[MTILE];
  } r;
};

// ---------------------------------------------------------------------------
// Prep: codebook f32 -> bf16 hi + bf16 lo residual, plus per-code ||c||^2.
// ---------------------------------------------------------------------------
__global__ __launch_bounds__(256) void vq_prep_kernel(const float* __restrict__ codebook,
                                                      __bf16* __restrict__ cb_hi,
                                                      __bf16* __restrict__ cb_lo,
                                                      float* __restrict__ cnorm) {
  const int k = blockIdx.x;
  const int d = threadIdx.x;
  float c = codebook[(size_t)k * KD + d];
  __bf16 h = (__bf16)c;
  __bf16 l = (__bf16)(c - (float)h);
  cb_hi[(size_t)k * KD + d] = h;
  cb_lo[(size_t)k * KD + d] = l;

  __shared__ float red[256];
  red[d] = c * c;
  __syncthreads();
  for (int s = 128; s > 0; s >>= 1) {
    if (d < s) red[d] += red[d + s];
    __syncthreads();
  }
  if (d == 0) cnorm[k] = red[0];
}

// ---------------------------------------------------------------------------
// Main: per block, 64 rows x all 2048 codes.  8 waves; wave w owns columns
// panel*128 + w*16 + [0,16).  3-pass split-bf16 WMMA, running argmin.
// ---------------------------------------------------------------------------
__global__ __launch_bounds__(256) void vq_main_kernel(const float* __restrict__ codes,
                                                      const float* __restrict__ codebook,
                                                      const __bf16* __restrict__ cb_hi,
                                                      const __bf16* __restrict__ cb_lo,
                                                      const float* __restrict__ cnorm,
                                                      float* __restrict__ out_nearest,
                                                      float* __restrict__ out_onehot) {
  __shared__ __attribute__((aligned(16))) SmemU sm;

  const int t     = threadIdx.x;
  const int wave  = t >> 5;
  const int lane  = t & 31;
  const int half  = lane >> 4;   // K-half selector per ISA WMMA layouts
  const int n     = lane & 15;   // row (A) / column (B) slot
  const int rbase = blockIdx.x * MTILE;

  // ---- Stage 64x256 f32 codes tile -> LDS as split bf16 (hi + residual) ----
  {
    const v4f* src = (const v4f*)(codes + (size_t)rbase * KD);
    for (int i = t; i < (MTILE * KD) / 4; i += 256) {
      v4f v = __builtin_nontemporal_load(&src[i]);   // read-once stream
      int row = i >> 6;            // 64 float4 per row
      int col = (i & 63) << 2;
      v4bf h4, l4;
#pragma unroll
      for (int j = 0; j < 4; ++j) {
        float x = v[j];
        __bf16 h = (__bf16)x;
        h4[j] = h;
        l4[j] = (__bf16)(x - (float)h);
      }
      *(v4bf*)&sm.g.a_hi[row][aswz(row, col)] = h4;
      *(v4bf*)&sm.g.a_lo[row][aswz(row, col)] = l4;
    }
  }
  __syncthreads();

  // ---- running per-lane argmin state: 4 row-tiles x 8 C rows ----
  float bestv[4][8];
  int   besti[4][8];
#pragma unroll
  for (int rt = 0; rt < 4; ++rt)
#pragma unroll
    for (int r = 0; r < 8; ++r) { bestv[rt][r] = 3.4e38f; besti[rt][r] = 0; }

  const v8f vzero = {0.f, 0.f, 0.f, 0.f, 0.f, 0.f, 0.f, 0.f};

#pragma unroll 1
  for (int panel = 0; panel < 16; ++panel) {
    const int col = panel * 128 + wave * 16 + n;  // this lane's B column
    const __bf16* bh_base = cb_hi + (size_t)col * KD + half * 16;
    const __bf16* bl_base = cb_lo + (size_t)col * KD + half * 16;

    v8f acc[4];
#pragma unroll
    for (int rt = 0; rt < 4; ++rt) acc[rt] = vzero;

    // unroll(1): keep LDS A-fragment loads adjacent to their WMMA uses so the
    // live set stays ~140 VGPRs (no scratch spills).
#pragma unroll 1
    for (int kb = 0; kb < KD; kb += 32) {
      // B fragments: lane n = column n; halves carry K 0-15 / 16-31 of chunk.
      v16bf bh = *(const v16bf*)(bh_base + kb);
      v16bf bl = *(const v16bf*)(bl_base + kb);
      __builtin_prefetch((const void*)(bh_base + kb + 32), 0, 3);

#pragma unroll
      for (int rt = 0; rt < 4; ++rt) {
        const int m = rt * 16 + n;  // A row
        // A layout: half 0 carries K 0-15, half 1 carries K 16-31 (matches B).
        v8bf ah0 = *(const v8bf*)&sm.g.a_hi[m][aswz(m, kb + half * 16)];
        v8bf ah1 = *(const v8bf*)&sm.g.a_hi[m][aswz(m, kb + half * 16 + 8)];
        v8bf al0 = *(const v8bf*)&sm.g.a_lo[m][aswz(m, kb + half * 16)];
        v8bf al1 = *(const v8bf*)&sm.g.a_lo[m][aswz(m, kb + half * 16 + 8)];
        v16bf ah = CAT16(ah0, ah1);
        v16bf al = CAT16(al0, al1);
        acc[rt] = wmma_bf16(ah, bh, acc[rt]);  // hi*hi
        acc[rt] = wmma_bf16(ah, bl, acc[rt]);  // hi*lo
        acc[rt] = wmma_bf16(al, bh, acc[rt]);  // lo*hi
      }
    }

    // score = ||c||^2 - 2 x.c   (||x||^2 dropped: constant per row)
    const float cn = cnorm[col];
#pragma unroll
    for (int rt = 0; rt < 4; ++rt)
#pragma unroll
      for (int r = 0; r < 8; ++r) {
        float s = cn - 2.0f * acc[rt][r];
        if (s < bestv[rt][r]) { bestv[rt][r] = s; besti[rt][r] = col; }
      }
  }

  __syncthreads();  // all waves done reading A tiles; LDS reused below

  // ---- reduce across the 16 column-slots of each half-wave via shuffles ----
#pragma unroll
  for (int rt = 0; rt < 4; ++rt)
#pragma unroll
    for (int r = 0; r < 8; ++r) {
      float v = bestv[rt][r];
      int   i = besti[rt][r];
#pragma unroll
      for (int m = 8; m >= 1; m >>= 1) {
        float ov = __shfl_xor(v, m, 16);
        int   oi = __shfl_xor(i, m, 16);
        if (ov < v || (ov == v && oi < i)) { v = ov; i = oi; }
      }
      if (n == 0) {
        int rowl = rt * 16 + half * 8 + r;  // C layout: VGPR r -> row r + 8*half
        sm.r.bestv[rowl][wave] = v;
        sm.r.besti[rowl][wave] = i;
      }
    }
  __syncthreads();

  // ---- final per-row reduce across the 8 waves ----
  if (t < MTILE) {
    float bv = sm.r.bestv[t][0];
    int   bi = sm.r.besti[t][0];
#pragma unroll
    for (int s = 1; s < 8; ++s) {
      float v = sm.r.bestv[t][s];
      int   i = sm.r.besti[t][s];
      if (v < bv || (v == bv && i < bi)) { bv = v; bi = i; }
    }
    sm.r.idxf[t] = bi;
  }
  __syncthreads();

  // ---- outputs (write-once streams: non-temporal stores) ----
  // quantized_st == nearest codebook row (straight-through is identity in fwd)
  for (int r = 0; r < MTILE; ++r) {
    int idx = sm.r.idxf[r];
    float v = codebook[(size_t)idx * KD + t];
    __builtin_nontemporal_store(v, &out_nearest[(size_t)(rbase + r) * KD + t]);
  }
  // one-hot rows: streamed 16B stores (dominant HBM traffic)
  v4f* oh = (v4f*)out_onehot;
  for (int r = 0; r < MTILE; ++r) {
    int idx = sm.r.idxf[r];
    size_t rowoff = (size_t)(rbase + r) * (NC / 4);
    for (int c4 = t; c4 < NC / 4; c4 += 256) {
      int base = c4 << 2;
      v4f v;
#pragma unroll
      for (int j = 0; j < 4; ++j) v[j] = (idx == base + j) ? 1.0f : 0.0f;
      __builtin_nontemporal_store(v, &oh[rowoff + c4]);
    }
  }
}

// ---------------------------------------------------------------------------
extern "C" void kernel_launch(void* const* d_in, const int* in_sizes, int n_in,
                              void* d_out, int out_size, void* d_ws, size_t ws_size,
                              hipStream_t stream) {
  (void)in_sizes; (void)n_in; (void)out_size; (void)ws_size;

  const float* codes    = (const float*)d_in[0];   // [32,1024,256] f32
  const float* codebook = (const float*)d_in[1];   // [2048,256] f32

  char* ws = (char*)d_ws;
  __bf16* cb_hi = (__bf16*)ws;                                        // 1 MB
  __bf16* cb_lo = (__bf16*)(ws + (size_t)NC * KD * sizeof(__bf16));   // 1 MB
  float*  cnorm = (float*)(ws + (size_t)2 * NC * KD * sizeof(__bf16)); // 8 KB

  float* out_nearest = (float*)d_out;                       // 32768*256 f32
  float* out_onehot  = out_nearest + (size_t)NROWS * KD;    // 32768*2048 f32

  vq_prep_kernel<<<NC, 256, 0, stream>>>(codebook, cb_hi, cb_lo, cnorm);
  vq_main_kernel<<<NROWS / MTILE, 256, 0, stream>>>(codes, codebook, cb_hi, cb_lo,
                                                    cnorm, out_nearest, out_onehot);
}
